// MiniCPM3Attention_41042707480910
// MI455X (gfx1250) — matrix operand order repaired and checked by reading, not compiled, not run on hardware
//
#include <hip/hip_runtime.h>
#include <hip/hip_bf16.h>
#include <stdint.h>

// ---------------------------------------------------------------------------
// MiniCPM3 MLA attention for gfx1250 (MI455X, wave32, WMMA).
// fp32 I/O, f16 WMMA compute with f32 accumulation; fp32 softmax.
// ---------------------------------------------------------------------------

#define HIDDEN    2560
#define N_HEADS   40
#define D_NOPE    64
#define D_ROPE    32
#define D_QK      96
#define D_V       64
#define Q_RANK    768
#define KV_RANK   256
#define KV_LAT    (KV_RANK + D_ROPE)   // 288
#define ROPE_LN_THETA 9.2103403719761836f  // ln(10000)
#define ATTN_SCALE 0.10206207261596575f    // 96^-0.5
#define EPS 1e-6f

typedef __attribute__((ext_vector_type(16))) _Float16 v16h;
typedef __attribute__((ext_vector_type(8)))  _Float16 v8h;
typedef __attribute__((ext_vector_type(8)))  float    v8f;

union Frag16 { v16h v; v8h h[2]; };

__device__ __forceinline__ v8f wmma_f16(v16h a, v16h b, v8f c) {
  return __builtin_amdgcn_wmma_f32_16x16x32_f16(
      /*neg_a=*/false, a, /*neg_b=*/false, b,
      /*c_mod=*/(short)0, c, /*reuse_a=*/false, /*reuse_b=*/false);
}

__device__ __forceinline__ v8h cvt8(float4 a, float4 b) {
  v8h h;
  h[0] = (_Float16)a.x; h[1] = (_Float16)a.y; h[2] = (_Float16)a.z; h[3] = (_Float16)a.w;
  h[4] = (_Float16)b.x; h[5] = (_Float16)b.y; h[6] = (_Float16)b.z; h[7] = (_Float16)b.w;
  return h;
}

// ---- Fragment loaders ------------------------------------------------------
// A-matrix 16x32 f16 (ISA 7.12.2): lane m = l&15; lanes 0-15 hold K=0..7
// (VGPR0-3) and 16..23 (VGPR4-7); lanes 16-31 hold K=8..15, 24..31.
__device__ __forceinline__ v16h lds_fragA(const _Float16* base, int ldh, int lane) {
  int m  = lane & 15;
  int kb = (lane >> 4) << 3;           // 0 or 8
  const _Float16* row = base + m * ldh;
  Frag16 f;
  f.h[0] = *(const v8h*)(row + kb);
  f.h[1] = *(const v8h*)(row + kb + 16);
  return f.v;
}
// B-matrix 32x16 f16: lane n = l&15; lanes 0-15 hold K=0..7 and 16..23,
// lanes 16-31 hold K=8..15, 24..31. LDS tile stored as [N][K].
__device__ __forceinline__ v16h lds_fragB(const _Float16* base, int ldh, int lane) {
  int n  = lane & 15;
  int kb = (lane >> 4) << 3;           // 0 or 8
  const _Float16* row = base + n * ldh;
  Frag16 f;
  f.h[0] = *(const v8h*)(row + kb);
  f.h[1] = *(const v8h*)(row + kb + 16);
  return f.v;
}
// A-layout fragment straight from fp32 global rows (vectorized b128 loads).
__device__ __forceinline__ v16h glb_fragA(const float* p, size_t stride,
                                          int row0, int kofs, int lane) {
  int r  = row0 + (lane & 15);
  int kb = (lane >> 4) << 3;
  const float* src = p + (size_t)r * stride + kofs + kb;
  float4 f0 = *(const float4*)(src);
  float4 f1 = *(const float4*)(src + 4);
  float4 g0 = *(const float4*)(src + 16);
  float4 g1 = *(const float4*)(src + 20);
  Frag16 f;
  f.h[0] = cvt8(f0, f1);
  f.h[1] = cvt8(g0, g1);
  return f.v;
}

// ---------------------------------------------------------------------------
// Generic GEMM: C[M,N] = A[M,K] * B[K,N], fp32 I/O, f16 WMMA compute.
// 256 threads (8 waves), 128x128 tile, K-step 32, double-buffered LDS
// staging (one barrier per K-step; global loads overlap WMMAs).
// Each wave computes 32M x 64N -> 8 WMMAs per K-step.
//
// Requirements (all launches below satisfy them):
//   M % 128 == 0, K % 32 == 0.
//   N_GUARD=false additionally requires N % 128 == 0.
//   N_GUARD=true  requires only N % 8 == 0 (OOB 8-chunks are fully OOB ->
//   zero-filled; no scalar fallback path is generated).
// ---------------------------------------------------------------------------
#define GT_M 128
#define GT_N 128
#define GT_K 32
#define LDSH 48   // LDS row stride in halfs (96B: 16B aligned, bank-spread)

template <bool N_GUARD>
__global__ void __launch_bounds__(256)
wmma_gemm_kernel(const float* __restrict__ A, const float* __restrict__ B,
                 float* __restrict__ C, int M, int N, int K) {
  __shared__ alignas(16) _Float16 As[2][GT_M * LDSH];
  __shared__ alignas(16) _Float16 Bs[2][GT_N * LDSH];   // transposed: [n][k]

  const int tid  = threadIdx.x;
  const int lane = tid & 31;
  const int wid  = tid >> 5;     // 0..7
  const int wm   = wid & 3;      // 4 waves over M (32 rows each)
  const int wn   = wid >> 2;     // 2 waves over N (64 cols each)
  const int m0   = blockIdx.y * GT_M;
  const int n0   = blockIdx.x * GT_N;

  v8f acc0[4] = {};
  v8f acc1[4] = {};

  // ---- staging helpers (branch-free fast paths) ----
  auto stageA = [&](int buf, int k0) {
    int i  = tid >> 1;             // row 0..127
    int jb = (tid & 1) << 4;       // 0 or 16
    const float* rowp = A + (size_t)(m0 + i) * K + k0;
    if (k0 + GT_K < K) __builtin_prefetch(rowp + GT_K + jb, 0, 1);
#pragma unroll
    for (int c = 0; c < 2; ++c) {
      int j = jb + c * 8;
      float4 f0 = *(const float4*)(rowp + j);
      float4 f1 = *(const float4*)(rowp + j + 4);
      *(v8h*)&As[buf][i * LDSH + j] = cvt8(f0, f1);
    }
  };
  auto stageB = [&](int buf, int k0) {
    int kk = tid >> 3;             // 0..31
    int jb = (tid & 7) << 4;       // 0..112
    const float* rowp = B + (size_t)(k0 + kk) * N + n0;
    if (k0 + GT_K < K) __builtin_prefetch(rowp + (size_t)GT_K * N + jb, 0, 1);
#pragma unroll
    for (int c = 0; c < 2; ++c) {
      int j = jb + c * 8;
      v8h hv = {};
      if (!N_GUARD || (n0 + j + 8 <= N)) {
        float4 f0 = *(const float4*)(rowp + j);
        float4 f1 = *(const float4*)(rowp + j + 4);
        hv = cvt8(f0, f1);
      }
#pragma unroll
      for (int q = 0; q < 8; ++q) Bs[buf][(j + q) * LDSH + kk] = hv[q];
    }
  };

  // ---- ping-pong main loop ----
  stageA(0, 0);
  stageB(0, 0);
  int p = 0;
  for (int k0 = 0; k0 < K; k0 += GT_K) {
    __syncthreads();   // staging of buf p visible; buf p^1 free to overwrite
    if (k0 + GT_K < K) {
      stageA(p ^ 1, k0 + GT_K);
      stageB(p ^ 1, k0 + GT_K);
    }
    v16h a0 = lds_fragA(&As[p][(wm * 32) * LDSH], LDSH, lane);
    v16h a1 = lds_fragA(&As[p][(wm * 32 + 16) * LDSH], LDSH, lane);
#pragma unroll
    for (int j = 0; j < 4; ++j) {
      v16h b = lds_fragB(&Bs[p][(wn * 64 + j * 16) * LDSH], LDSH, lane);
      acc0[j] = wmma_f16(a0, b, acc0[j]);
      acc1[j] = wmma_f16(a1, b, acc1[j]);
    }
    p ^= 1;
  }

  // C/D layout: lane n = l&15, row m = vgpr + 8*(l>>4).
  const int nlo = lane & 15;
  const int mb  = (lane >> 4) << 3;
#pragma unroll
  for (int j = 0; j < 4; ++j) {
    int cn = n0 + wn * 64 + j * 16 + nlo;
    if (!N_GUARD || cn < N) {
#pragma unroll
      for (int v = 0; v < 8; ++v) {
        int r0 = m0 + wm * 32 + mb + v;
        C[(size_t)r0 * N + cn]        = acc0[j][v];
        C[(size_t)(r0 + 16) * N + cn] = acc1[j][v];
      }
    }
  }
}

// ---------------------------------------------------------------------------
// RMSNorm: y[row] = x[row] * rsqrt(mean(x^2)+eps) * w   (one block per row)
// ---------------------------------------------------------------------------
__global__ void __launch_bounds__(256)
rmsnorm_kernel(const float* __restrict__ x, const float* __restrict__ w,
               float* __restrict__ y, int D, int in_stride, int out_stride) {
  __shared__ float red[256];
  const int row = blockIdx.x;
  const float* xr = x + (size_t)row * in_stride;
  float s = 0.0f;
  for (int i = threadIdx.x; i < D; i += 256) { float v = xr[i]; s += v * v; }
  red[threadIdx.x] = s;
  __syncthreads();
  for (int off = 128; off > 0; off >>= 1) {
    if (threadIdx.x < off) red[threadIdx.x] += red[threadIdx.x + off];
    __syncthreads();
  }
  const float r = rsqrtf(red[0] / (float)D + EPS);
  float* yr = y + (size_t)row * out_stride;
  for (int i = threadIdx.x; i < D; i += 256) yr[i] = xr[i] * r * w[i];
}

// ---------------------------------------------------------------------------
// Pack kernels: head-major Q/K (NeoX RoPE on last 32 dims) and V^T.
// ---------------------------------------------------------------------------
__device__ __forceinline__ void rope_cs(int pos, int i, float* c, float* s) {
  float inv = __expf(-((float)i / 16.0f) * ROPE_LN_THETA);  // 10000^(-2i/32)
  float f = (float)pos * inv;
  *c = __cosf(f);
  *s = __sinf(f);
}

__global__ void __launch_bounds__(256)
build_q_full_kernel(const float* __restrict__ qb, const int* __restrict__ pos,
                    float* __restrict__ qfull, int T) {
  size_t e = (size_t)blockIdx.x * 256 + threadIdx.x;
  size_t total = (size_t)T * N_HEADS * D_QK;
  if (e >= total) return;
  int t = (int)(e / (N_HEADS * D_QK));
  int r = (int)(e % (N_HEADS * D_QK));
  int h = r / D_QK;
  int d = r % D_QK;
  const float* src = qb + (size_t)t * (N_HEADS * D_QK) + h * D_QK;
  float val;
  if (d < D_NOPE) {
    val = src[d];
  } else if (d < D_NOPE + 16) {
    int i = d - D_NOPE;
    float c, s; rope_cs(pos[t], i, &c, &s);
    val = src[D_NOPE + i] * c - src[D_NOPE + 16 + i] * s;
  } else {
    int i = d - D_NOPE - 16;
    float c, s; rope_cs(pos[t], i, &c, &s);
    val = src[D_NOPE + 16 + i] * c + src[D_NOPE + i] * s;
  }
  qfull[((size_t)h * T + t) * D_QK + d] = val;
}

__global__ void __launch_bounds__(256)
build_k_full_kernel(const float* __restrict__ kv, const float* __restrict__ latent,
                    const int* __restrict__ pos, float* __restrict__ kfull, int T) {
  size_t e = (size_t)blockIdx.x * 256 + threadIdx.x;
  size_t total = (size_t)T * N_HEADS * D_QK;
  if (e >= total) return;
  int t = (int)(e / (N_HEADS * D_QK));
  int r = (int)(e % (N_HEADS * D_QK));
  int h = r / D_QK;
  int d = r % D_QK;
  float val;
  if (d < D_NOPE) {
    val = kv[(size_t)t * (N_HEADS * (D_NOPE + D_V)) + h * (D_NOPE + D_V) + d];
  } else {
    const float* pe = latent + (size_t)t * KV_LAT + KV_RANK;  // 32 rope dims
    if (d < D_NOPE + 16) {
      int i = d - D_NOPE;
      float c, s; rope_cs(pos[t], i, &c, &s);
      val = pe[i] * c - pe[16 + i] * s;
    } else {
      int i = d - D_NOPE - 16;
      float c, s; rope_cs(pos[t], i, &c, &s);
      val = pe[16 + i] * c + pe[i] * s;
    }
  }
  kfull[((size_t)h * T + t) * D_QK + d] = val;
}

__global__ void __launch_bounds__(256)
build_v_t_kernel(const float* __restrict__ kv, float* __restrict__ vt, int T) {
  size_t e = (size_t)blockIdx.x * 256 + threadIdx.x;
  size_t total = (size_t)N_HEADS * D_V * T;
  if (e >= total) return;
  int h = (int)(e / ((size_t)D_V * T));
  int r = (int)(e % ((size_t)D_V * T));
  int d = r / T;
  int t = r % T;
  vt[((size_t)h * D_V + d) * T + t] =
      kv[(size_t)t * (N_HEADS * (D_NOPE + D_V)) + h * (D_NOPE + D_V) + D_NOPE + d];
}

// ---------------------------------------------------------------------------
// Flash attention: 4 waves per block; block covers (head, 64 query rows).
// K (32x96) and V^T (64x32) tiles staged cooperatively into LDS per key step;
// each wave owns 16 query rows + private online softmax + private P tile.
// ---------------------------------------------------------------------------
#define KS_LD 104   // Ks row stride in halfs (208B)
#define VS_LD 40    // Vs row stride in halfs (80B)

__global__ void __launch_bounds__(128)
flash_attn_kernel(const float* __restrict__ qf,  // [H][T][96]
                  const float* __restrict__ kf,  // [H][T][96]
                  const float* __restrict__ vt,  // [H][64][T]
                  float* __restrict__ out,       // [T][H*64]
                  int T) {
  __shared__ alignas(16) _Float16 Ks[32 * KS_LD];
  __shared__ alignas(16) _Float16 Vs[64 * VS_LD];
  __shared__ alignas(16) _Float16 Ps[4][16 * 32];

  const int h    = blockIdx.y;
  const int tb   = blockIdx.x * 64;
  const int tid  = threadIdx.x;
  const int lane = tid & 31;
  const int wid  = tid >> 5;
  const int t0   = tb + wid * 16;
  const float* Q  = qf + (size_t)h * T * D_QK;
  const float* Kp = kf + (size_t)h * T * D_QK;
  const float* Vt = vt + (size_t)h * D_V * T;

  // Per-wave Q fragments (K=96 -> 3 chunks of 32), kept in VGPRs.
  v16h qa[3];
#pragma unroll
  for (int c = 0; c < 3; ++c) qa[c] = glb_fragA(Q, D_QK, t0, c * 32, lane);

  const int ncol = lane & 15;
  const int mb   = (lane >> 4) << 3;

  float rm[8], rl[8];
#pragma unroll
  for (int v = 0; v < 8; ++v) { rm[v] = -1e30f; rl[v] = 0.0f; }
  v8f o[4] = {};

  for (int sb = 0; sb < tb + 64; sb += 32) {
    // ---- cooperative staging: K rows sb..sb+31 (96 f32 each -> f16) ----
    {
      int i  = tid >> 2;              // key row 0..31
      int jb = (tid & 3) << 3;        // 0,8,16,24
      const float* src = Kp + (size_t)(sb + i) * D_QK;
#pragma unroll
      for (int c = 0; c < 3; ++c) {
        int j = jb + c * 32;
        float4 f0 = *(const float4*)(src + j);
        float4 f1 = *(const float4*)(src + j + 4);
        *(v8h*)&Ks[i * KS_LD + j] = cvt8(f0, f1);
      }
    }
    // ---- cooperative staging: V^T rows d=0..63, cols sb..sb+31 ----
    {
      int d  = tid >> 1;              // 0..63
      int jb = (tid & 1) << 4;        // 0 or 16
      const float* src = Vt + (size_t)d * T + sb;
#pragma unroll
      for (int c = 0; c < 2; ++c) {
        int j = jb + c * 8;
        float4 f0 = *(const float4*)(src + j);
        float4 f1 = *(const float4*)(src + j + 4);
        *(v8h*)&Vs[d * VS_LD + j] = cvt8(f0, f1);
      }
    }
    __syncthreads();

    // ---- scores: two 16x16 tiles over keys sb..sb+31 ----
    v8f sa0 = {};
    v8f sa1 = {};
#pragma unroll
    for (int c = 0; c < 3; ++c) {
      v16h kb0 = lds_fragB(&Ks[c * 32], KS_LD, lane);
      v16h kb1 = lds_fragB(&Ks[16 * KS_LD + c * 32], KS_LD, lane);
      sa0 = wmma_f16(qa[c], kb0, sa0);
      sa1 = wmma_f16(qa[c], kb1, sa1);
    }

    // ---- fp32 online softmax (rows m = mb+v, cols = lane&15 per tile) ----
    float corr[8];
#pragma unroll
    for (int v = 0; v < 8; ++v) {
      const int t = t0 + mb + v;
      float x0 = sa0[v] * ATTN_SCALE;
      float x1 = sa1[v] * ATTN_SCALE;
      if (sb + ncol > t)      x0 = -1e30f;
      if (sb + 16 + ncol > t) x1 = -1e30f;
      float mx = fmaxf(x0, x1);
#pragma unroll
      for (int off = 1; off < 16; off <<= 1)
        mx = fmaxf(mx, __shfl_xor(mx, off, 32));
      const float nm = fmaxf(rm[v], mx);
      corr[v] = __expf(rm[v] - nm);
      rm[v] = nm;
      const float p0 = __expf(x0 - nm);
      const float p1 = __expf(x1 - nm);
      float ps = p0 + p1;
#pragma unroll
      for (int off = 1; off < 16; off <<= 1)
        ps += __shfl_xor(ps, off, 32);
      rl[v] = rl[v] * corr[v] + ps;
      Ps[wid][(mb + v) * 32 + ncol]      = (_Float16)p0;
      Ps[wid][(mb + v) * 32 + 16 + ncol] = (_Float16)p1;
    }
#pragma unroll
    for (int d = 0; d < 4; ++d)
#pragma unroll
      for (int v = 0; v < 8; ++v) o[d][v] *= corr[v];

    // Cross-lane LDS dep within the wave: drain DS stores before reads.
    asm volatile("s_wait_dscnt 0" ::: "memory");

    // ---- O += P V : P as A-fragment, shared V^T rows as B-fragment ----
    v16h pa = lds_fragA(Ps[wid], 32, lane);
#pragma unroll
    for (int d = 0; d < 4; ++d) {
      v16h vb = lds_fragB(&Vs[(d * 16) * VS_LD], VS_LD, lane);
      o[d] = wmma_f16(pa, vb, o[d]);
    }
    __syncthreads();
  }

  // ---- finalize: divide by softmax denominator, write [t][h*64+d] ----
#pragma unroll
  for (int d = 0; d < 4; ++d) {
#pragma unroll
    for (int v = 0; v < 8; ++v) {
      const int t = t0 + mb + v;
      out[(size_t)t * (N_HEADS * D_V) + h * D_V + d * 16 + ncol] = o[d][v] / rl[v];
    }
  }
}

// ---------------------------------------------------------------------------
// Host-side launcher
// ---------------------------------------------------------------------------
static inline int cdiv(int a, int b) { return (a + b - 1) / b; }

static void launch_gemm(const float* A, const float* B, float* C,
                        int M, int N, int K, hipStream_t stream) {
  dim3 grid(cdiv(N, GT_N), cdiv(M, GT_M));
  if (N % GT_N == 0)
    wmma_gemm_kernel<false><<<grid, 256, 0, stream>>>(A, B, C, M, N, K);
  else
    wmma_gemm_kernel<true><<<grid, 256, 0, stream>>>(A, B, C, M, N, K);
}

extern "C" void kernel_launch(void* const* d_in, const int* in_sizes, int n_in,
                              void* d_out, int out_size, void* d_ws, size_t ws_size,
                              hipStream_t stream) {
  const int*   positions = (const int*)  d_in[0];
  const float* hidden    = (const float*)d_in[1];
  const float* q_a_w     = (const float*)d_in[2];
  const float* q_a_ln    = (const float*)d_in[3];
  const float* q_b_w     = (const float*)d_in[4];
  const float* kv_a_w    = (const float*)d_in[5];
  const float* kv_a_ln   = (const float*)d_in[6];
  const float* kv_b_w    = (const float*)d_in[7];
  const float* o_w       = (const float*)d_in[8];
  float*       out       = (float*)d_out;

  const int T = in_sizes[0];  // 2048

  // Workspace layout (fp32 elements), total ~196 MB.
  float* ws = (float*)d_ws;
  size_t off = 0;
  float* q_lat  = ws + off; off += (size_t)T * Q_RANK;
  float* q_n    = ws + off; off += (size_t)T * Q_RANK;
  float* qb     = ws + off; off += (size_t)T * N_HEADS * D_QK;
  float* latent = ws + off; off += (size_t)T * KV_LAT;
  float* kva_n  = ws + off; off += (size_t)T * KV_RANK;
  float* kv     = ws + off; off += (size_t)T * N_HEADS * (D_NOPE + D_V);
  float* qfull  = ws + off; off += (size_t)N_HEADS * T * D_QK;
  float* kfull  = ws + off; off += (size_t)N_HEADS * T * D_QK;
  float* vt     = ws + off; off += (size_t)N_HEADS * D_V * T;
  float* attnf  = ws + off; off += (size_t)T * N_HEADS * D_V;

  // 1. q_lat = H @ q_a_w                       [T,2560]x[2560,768]
  launch_gemm(hidden, q_a_w, q_lat, T, Q_RANK, HIDDEN, stream);
  // 2. q_n = rmsnorm(q_lat) * q_a_ln
  rmsnorm_kernel<<<T, 256, 0, stream>>>(q_lat, q_a_ln, q_n, Q_RANK, Q_RANK, Q_RANK);
  // 3. qb = q_n @ q_b_w                        [T,768]x[768,3840]
  launch_gemm(q_n, q_b_w, qb, T, N_HEADS * D_QK, Q_RANK, stream);
  // 4. latent = H @ kv_a_w                     [T,2560]x[2560,288]  (N-guarded)
  launch_gemm(hidden, kv_a_w, latent, T, KV_LAT, HIDDEN, stream);
  // 5. kva_n = rmsnorm(latent[:, :256]) * kv_a_ln
  rmsnorm_kernel<<<T, 256, 0, stream>>>(latent, kv_a_ln, kva_n, KV_RANK, KV_LAT, KV_RANK);
  // 6. kv = kva_n @ kv_b_w                     [T,256]x[256,5120]
  launch_gemm(kva_n, kv_b_w, kv, T, N_HEADS * (D_NOPE + D_V), KV_RANK, stream);

  // 7. pack head-major tensors (+RoPE).
  {
    size_t tot = (size_t)T * N_HEADS * D_QK;
    int g = (int)((tot + 255) / 256);
    build_q_full_kernel<<<g, 256, 0, stream>>>(qb, positions, qfull, T);
    build_k_full_kernel<<<g, 256, 0, stream>>>(kv, latent, positions, kfull, T);
    size_t tv = (size_t)N_HEADS * D_V * T;
    build_v_t_kernel<<<(int)((tv + 255) / 256), 256, 0, stream>>>(kv, vt, T);
  }

  // 8. flash attention: 4 waves / block, 64 query rows per block.
  flash_attn_kernel<<<dim3(T / 64, N_HEADS), 128, 0, stream>>>(qfull, kfull, vt, attnf, T);

  // 9. out = attnf @ o_w                       [T,2560]x[2560,2560]
  launch_gemm(attnf, o_w, out, T, HIDDEN, N_HEADS * D_V, stream);
}
